// EZGNN_58780922413737
// MI455X (gfx1250) — matrix operand order repaired and checked by reading, not compiled, not run on hardware
//
#include <hip/hip_runtime.h>
#include <hip/hip_bf16.h>
#include <math.h>

typedef __attribute__((ext_vector_type(16))) _Float16 v16h;
typedef __attribute__((ext_vector_type(8)))  _Float16 v8h;
typedef __attribute__((ext_vector_type(8)))  float    v8f;

#define NE      1560
#define NN      40
#define SIGL    32768
#define NWIN    8
#define WIN     1024
#define STRW    4096
#define HALO    16
#define PT      (WIN + 2*HALO)   // 1056 padded timesteps

// ---------------------------------------------------------------------------
// Kernel 1: TCN over one (edge, window). 8 waves, WMMA for layers 2-4.
// ---------------------------------------------------------------------------
__global__ __launch_bounds__(256) void EZGNN_tcn_kernel(
    const float* __restrict__ sig, const int* __restrict__ ei,
    const float* __restrict__ w1, const float* __restrict__ b1,
    const float* __restrict__ w2, const float* __restrict__ b2,
    const float* __restrict__ w3, const float* __restrict__ b3,
    const float* __restrict__ w4, const float* __restrict__ b4,
    const float* __restrict__ qv,
    float* __restrict__ zwin, float* __restrict__ logits)
{
  __shared__ __align__(32) _Float16 bufA[PT * 16];
  __shared__ __align__(32) _Float16 bufB[PT * 16];
  __shared__ float dbuf[WIN + 4];
  __shared__ float sw1[80];
  __shared__ float sb1[16];
  __shared__ float sbl[48];     // biases of layers 2,3,4
  __shared__ float red[256];
  __shared__ float zsh[16];

  const int tid = threadIdx.x;
  const int bid = blockIdx.x;           // e*8 + w
  const int e = bid >> 3;
  const int w = bid & 7;
  const int srcn = ei[e];
  const int dstn = ei[NE + e];

  // stage small params
  if (tid < 80) sw1[tid] = w1[tid];
  if (tid < 16) {
    sb1[tid] = b1[tid];
    sbl[tid] = b2[tid];
    sbl[16 + tid] = b3[tid];
    sbl[32 + tid] = b4[tid];
  }
  // zero halos of both ping-pong buffers (interior is always overwritten;
  // front halo doubles as the guaranteed-zero B fragment for the missing tap)
  {
    int i = tid;  // HALO*16 == 256 == blockDim
    bufA[i] = (_Float16)0.f;  bufB[i] = (_Float16)0.f;
    bufA[(HALO + WIN) * 16 + i] = (_Float16)0.f;
    bufB[(HALO + WIN) * 16 + i] = (_Float16)0.f;
  }
  // diff window into f32 buffer (pad 2 each side for layer 1)
  {
    const float* ps = sig + (size_t)srcn * SIGL + (size_t)w * STRW;
    const float* pd = sig + (size_t)dstn * SIGL + (size_t)w * STRW;
    for (int i = 0; i < 4; ++i) {
      int t = tid + 256 * i;
      dbuf[2 + t] = ps[t] - pd[t];
    }
    if (tid < 2) { dbuf[tid] = 0.f; dbuf[WIN + 2 + tid] = 0.f; }
  }
  __syncthreads();

  // ---- layer 1: conv 1->16, dil 1, pad 2 (plain VALU) ----
  for (int i = 0; i < 4; ++i) {
    int t = tid + 256 * i;
    float d0 = dbuf[t], d1 = dbuf[t + 1], d2 = dbuf[t + 2],
          d3 = dbuf[t + 3], d4 = dbuf[t + 4];
    v8h lo, hi;
    #pragma unroll
    for (int oc = 0; oc < 16; ++oc) {
      float a = sb1[oc] + sw1[oc*5+0]*d0 + sw1[oc*5+1]*d1 + sw1[oc*5+2]*d2
                        + sw1[oc*5+3]*d3 + sw1[oc*5+4]*d4;
      a = a > 0.f ? a : 0.f;
      if (oc < 8) lo[oc] = (_Float16)a; else hi[oc - 8] = (_Float16)a;
    }
    *(v8h*)&bufA[(t + HALO) * 16]     = lo;
    *(v8h*)&bufA[(t + HALO) * 16 + 8] = hi;
  }
  __syncthreads();

  // ---- layers 2..4: conv 16->16, dil 2/4/8, via v_wmma_f32_16x16x32_f16 ----
  const int lane = tid & 31;
  const int wave = tid >> 5;
  const int mrow = lane & 15;            // A row / B column / time-within-tile
  const int icb  = (lane >> 4) * 8;      // 0 or 8
  const float* Wg[3] = { w2, w3, w4 };
  const int dils[3]  = { 2, 4, 8 };

  _Float16* inb  = bufA;
  _Float16* outb = bufB;

  for (int l = 0; l < 3; ++l) {
    // Build A fragments: pair p packs taps (2p, 2p+1) into K=32.
    // Lane layout (16-bit A): lane m: tap 2p, ic 0-15; lane m+16: tap 2p+1, ic 0-15 (matches B).
    v16h afr[3];
    const float* Wl = Wg[l];
    #pragma unroll
    for (int p = 0; p < 3; ++p) {
      #pragma unroll
      for (int j = 0; j < 16; ++j) {
        int k  = (lane < 16) ? 2 * p : 2 * p + 1;
        int ic = j;
        afr[p][j] = (k <= 4) ? (_Float16)Wl[mrow * 80 + ic * 5 + k]
                             : (_Float16)0.f;
      }
    }
    const int dil = dils[l];
    // Per-lane per-pair time offset for B loads. Invalid tap (k==5, pair 2,
    // high lanes) reads the always-zero front halo instead (branchless).
    int  tsoff[3];
    bool tval[3];
    #pragma unroll
    for (int p = 0; p < 3; ++p) {
      int k = (lane < 16) ? 2 * p : 2 * p + 1;
      tval[p]  = (k <= 4);
      tsoff[p] = (k - 2) * dil + HALO;
    }
    // Two independent tiles per iteration -> two independent WMMA chains
    // that hide the WMMA->WMMA hazard latency.
    for (int tile = wave; tile < 64; tile += 16) {
      const int t0a = tile * 16;
      const int t0b = (tile + 8) * 16;
      v8f c0 = {}, c1 = {};
      #pragma unroll
      for (int p = 0; p < 3; ++p) {
        int tsa = tval[p] ? (t0a + mrow + tsoff[p]) : mrow;
        int tsb = tval[p] ? (t0b + mrow + tsoff[p]) : mrow;
        v16h bf0 = *(const v16h*)&inb[tsa * 16];
        v16h bf1 = *(const v16h*)&inb[tsb * 16];
        c0 = __builtin_amdgcn_wmma_f32_16x16x32_f16(
            false, afr[p], false, bf0, (short)0, c0, false, false);
        c1 = __builtin_amdgcn_wmma_f32_16x16x32_f16(
            false, afr[p], false, bf1, (short)0, c1, false, false);
      }
      // epilogue: bias + relu, pack 8 contiguous channels, one 16B store each
      const int tta = t0a + mrow + HALO;
      const int ttb = t0b + mrow + HALO;
      v8h ov0, ov1;
      #pragma unroll
      for (int r = 0; r < 8; ++r) {
        float a0 = c0[r] + sbl[l * 16 + icb + r];
        float a1 = c1[r] + sbl[l * 16 + icb + r];
        ov0[r] = (_Float16)(a0 > 0.f ? a0 : 0.f);
        ov1[r] = (_Float16)(a1 > 0.f ? a1 : 0.f);
      }
      *(v8h*)&outb[tta * 16 + icb] = ov0;
      *(v8h*)&outb[ttb * 16 + icb] = ov1;
    }
    __syncthreads();
    _Float16* tmp = inb; inb = outb; outb = tmp;
  }
  // final layer-4 output now in `inb`

  // ---- max over time per channel ----
  {
    int ch = tid & 15, grp = tid >> 4;
    float mx = -3.4e38f;
    for (int t = grp; t < WIN; t += 16) {
      float v = (float)inb[(t + HALO) * 16 + ch];
      mx = mx > v ? mx : v;
    }
    red[grp * 16 + ch] = mx;
  }
  __syncthreads();
  if (tid < 16) {
    float mm = red[tid];
    #pragma unroll
    for (int g = 1; g < 16; ++g) {
      float v = red[g * 16 + tid];
      mm = mm > v ? mm : v;
    }
    zwin[(size_t)bid * 16 + tid] = mm;
    zsh[tid] = mm;
  }
  __syncthreads();
  if (tid == 0) {
    float lg = 0.f;
    #pragma unroll
    for (int d = 0; d < 16; ++d) lg += zsh[d] * qv[d];
    logits[bid] = lg;
  }
}

// ---------------------------------------------------------------------------
// Kernel 2: per-edge attention softmax over windows + edge MLP
// ---------------------------------------------------------------------------
__global__ void EZGNN_edge_attn_kernel(
    const float* __restrict__ zwin, const float* __restrict__ logits,
    const float* __restrict__ eW, const float* __restrict__ eb,
    float* __restrict__ edge_attr)
{
  int e = blockIdx.x * blockDim.x + threadIdx.x;
  if (e >= NE) return;
  float lg[NWIN];
  float mx = -3.4e38f;
  #pragma unroll
  for (int w = 0; w < NWIN; ++w) { lg[w] = logits[e * NWIN + w]; mx = mx > lg[w] ? mx : lg[w]; }
  float s = 0.f;
  #pragma unroll
  for (int w = 0; w < NWIN; ++w) { lg[w] = __expf(lg[w] - mx); s += lg[w]; }
  float inv = 1.f / s;
  float ze[16];
  #pragma unroll
  for (int d = 0; d < 16; ++d) ze[d] = 0.f;
  for (int w = 0; w < NWIN; ++w) {
    float a = lg[w] * inv;
    const float* zp = zwin + ((size_t)e * NWIN + w) * 16;
    #pragma unroll
    for (int d = 0; d < 16; ++d) ze[d] += a * zp[d];
  }
  for (int h = 0; h < 16; ++h) {
    float a = eb[h];
    #pragma unroll
    for (int d = 0; d < 16; ++d) a += eW[h * 16 + d] * ze[d];
    edge_attr[e * 16 + h] = a;
  }
}

// ---------------------------------------------------------------------------
// Kernel 3: node embedding hx = x @ node_W^T + node_b
// ---------------------------------------------------------------------------
__global__ void EZGNN_node_init_kernel(
    const float* __restrict__ x, const float* __restrict__ nW,
    const float* __restrict__ nb, float* __restrict__ hx)
{
  int i = blockIdx.x * blockDim.x + threadIdx.x;
  if (i >= NN * 16) return;
  int n = i >> 4, h = i & 15;
  float a = nb[h];
  #pragma unroll
  for (int d = 0; d < 3; ++d) a += nW[h * 3 + d] * x[n * 3 + d];
  hx[i] = a;
}

// ---------------------------------------------------------------------------
// Kernel 4a: per-edge GNN message MLP + layernorm (layer l)
// ---------------------------------------------------------------------------
__global__ void EZGNN_gnn_msg_kernel(
    const float* __restrict__ hx, const float* __restrict__ ea,
    const int* __restrict__ ei,
    const float* __restrict__ Wi, const float* __restrict__ bi,
    const float* __restrict__ Wj, const float* __restrict__ bj,
    const float* __restrict__ We, const float* __restrict__ be,
    const float* __restrict__ m1W, const float* __restrict__ m1b,
    const float* __restrict__ m2W, const float* __restrict__ m2b,
    const float* __restrict__ lng, const float* __restrict__ lnb,
    int l, float* __restrict__ mout)
{
  int e = blockIdx.x * blockDim.x + threadIdx.x;
  if (e >= NE) return;
  const float* wi  = Wi  + l * 256;  const float* vbi = bi  + l * 16;
  const float* wj  = Wj  + l * 256;  const float* vbj = bj  + l * 16;
  const float* we  = We  + l * 256;  const float* vbe = be  + l * 16;
  const float* w1m = m1W + l * 16 * 48; const float* vb1 = m1b + l * 16;
  const float* w2m = m2W + l * 256;  const float* vb2 = m2b + l * 16;
  const float* g   = lng + l * 16;   const float* bb  = lnb + l * 16;

  int sn = ei[e], dn = ei[NE + e];
  float xi[16], xj[16], ev[16], mv[48];
  #pragma unroll
  for (int h = 0; h < 16; ++h) {
    xi[h] = hx[dn * 16 + h];
    xj[h] = hx[sn * 16 + h];
    ev[h] = ea[e * 16 + h];
  }
  for (int h = 0; h < 16; ++h) {
    float a = vbi[h], b = vbj[h], c = vbe[h];
    #pragma unroll
    for (int k = 0; k < 16; ++k) {
      a += wi[h * 16 + k] * xi[k];
      b += wj[h * 16 + k] * xj[k];
      c += we[h * 16 + k] * ev[k];
    }
    mv[h] = a; mv[16 + h] = b; mv[32 + h] = c;
  }
  float hid[16];
  for (int h = 0; h < 16; ++h) {
    float a = vb1[h];
    #pragma unroll
    for (int k = 0; k < 48; ++k) a += w1m[h * 48 + k] * mv[k];
    hid[h] = a > 0.f ? a : 0.f;
  }
  float o[16];
  float mu = 0.f;
  for (int h = 0; h < 16; ++h) {
    float a = vb2[h];
    #pragma unroll
    for (int k = 0; k < 16; ++k) a += w2m[h * 16 + k] * hid[k];
    o[h] = a; mu += a;
  }
  mu *= (1.f / 16.f);
  float var = 0.f;
  #pragma unroll
  for (int h = 0; h < 16; ++h) { float d = o[h] - mu; var += d * d; }
  var *= (1.f / 16.f);
  float inv = rsqrtf(var + 1e-5f);
  #pragma unroll
  for (int h = 0; h < 16; ++h)
    mout[e * 16 + h] = (o[h] - mu) * inv * g[h] + bb[h];
}

// ---------------------------------------------------------------------------
// Kernel 4b: deterministic segment-sum by dst + residual relu
// ---------------------------------------------------------------------------
__global__ void EZGNN_gnn_aggr_kernel(
    const float* __restrict__ mout, const int* __restrict__ ei,
    float* __restrict__ hx)
{
  int i = blockIdx.x * blockDim.x + threadIdx.x;
  if (i >= NN * 16) return;
  int n = i >> 4, h = i & 15;
  float s = 0.f;
  for (int e = 0; e < NE; ++e)
    if (ei[NE + e] == n) s += mout[e * 16 + h];
  float v = hx[i] + s;
  hx[i] = v > 0.f ? v : 0.f;
}

// ---------------------------------------------------------------------------
// Kernel 5: head (layernorm -> 32 relu -> 1)
// ---------------------------------------------------------------------------
__global__ void EZGNN_head_kernel(
    const float* __restrict__ hx,
    const float* __restrict__ g, const float* __restrict__ bb,
    const float* __restrict__ W1, const float* __restrict__ b1,
    const float* __restrict__ W2, const float* __restrict__ b2,
    float* __restrict__ out)
{
  int n = blockIdx.x * blockDim.x + threadIdx.x;
  if (n >= NN) return;
  float o[16];
  float mu = 0.f;
  #pragma unroll
  for (int h = 0; h < 16; ++h) { o[h] = hx[n * 16 + h]; mu += o[h]; }
  mu *= (1.f / 16.f);
  float var = 0.f;
  #pragma unroll
  for (int h = 0; h < 16; ++h) { float d = o[h] - mu; var += d * d; }
  var *= (1.f / 16.f);
  float inv = rsqrtf(var + 1e-5f);
  #pragma unroll
  for (int h = 0; h < 16; ++h) o[h] = (o[h] - mu) * inv * g[h] + bb[h];
  float acc = b2[0];
  for (int h = 0; h < 32; ++h) {
    float a = b1[h];
    #pragma unroll
    for (int d = 0; d < 16; ++d) a += W1[h * 16 + d] * o[d];
    a = a > 0.f ? a : 0.f;
    acc += W2[h] * a;
  }
  out[n] = acc;
}

// ---------------------------------------------------------------------------
extern "C" void kernel_launch(void* const* d_in, const int* in_sizes, int n_in,
                              void* d_out, int out_size, void* d_ws, size_t ws_size,
                              hipStream_t stream) {
  const float* x        = (const float*)d_in[0];
  const float* signals  = (const float*)d_in[1];
  const int*   ei       = (const int*)  d_in[2];
  const float* tcn_w1   = (const float*)d_in[3];
  const float* tcn_b1   = (const float*)d_in[4];
  const float* tcn_w2   = (const float*)d_in[5];
  const float* tcn_b2   = (const float*)d_in[6];
  const float* tcn_w3   = (const float*)d_in[7];
  const float* tcn_b3   = (const float*)d_in[8];
  const float* tcn_w4   = (const float*)d_in[9];
  const float* tcn_b4   = (const float*)d_in[10];
  const float* q_attn   = (const float*)d_in[11];
  const float* node_W   = (const float*)d_in[12];
  const float* node_b   = (const float*)d_in[13];
  const float* edge_W   = (const float*)d_in[14];
  const float* edge_b   = (const float*)d_in[15];
  const float* gnn_Wi   = (const float*)d_in[16];
  const float* gnn_bi   = (const float*)d_in[17];
  const float* gnn_Wj   = (const float*)d_in[18];
  const float* gnn_bj   = (const float*)d_in[19];
  const float* gnn_We   = (const float*)d_in[20];
  const float* gnn_be   = (const float*)d_in[21];
  const float* gnn_m1W  = (const float*)d_in[22];
  const float* gnn_m1b  = (const float*)d_in[23];
  const float* gnn_m2W  = (const float*)d_in[24];
  const float* gnn_m2b  = (const float*)d_in[25];
  const float* gnn_lng  = (const float*)d_in[26];
  const float* gnn_lnb  = (const float*)d_in[27];
  const float* head_lng = (const float*)d_in[28];
  const float* head_lnb = (const float*)d_in[29];
  const float* head_W1  = (const float*)d_in[30];
  const float* head_b1  = (const float*)d_in[31];
  const float* head_W2  = (const float*)d_in[32];
  const float* head_b2  = (const float*)d_in[33];

  // workspace layout (floats)
  float* ws        = (float*)d_ws;
  float* zwin      = ws;                        // NE*8*16
  float* logits    = zwin + NE * NWIN * 16;     // NE*8
  float* edge_attr = logits + NE * NWIN;        // NE*16
  float* hx        = edge_attr + NE * 16;       // NN*16
  float* mbuf      = hx + NN * 16;              // NE*16

  EZGNN_tcn_kernel<<<NE * NWIN, 256, 0, stream>>>(
      signals, ei, tcn_w1, tcn_b1, tcn_w2, tcn_b2, tcn_w3, tcn_b3,
      tcn_w4, tcn_b4, q_attn, zwin, logits);

  EZGNN_edge_attn_kernel<<<(NE + 255) / 256, 256, 0, stream>>>(
      zwin, logits, edge_W, edge_b, edge_attr);

  EZGNN_node_init_kernel<<<(NN * 16 + 255) / 256, 256, 0, stream>>>(
      x, node_W, node_b, hx);

  for (int l = 0; l < 4; ++l) {
    EZGNN_gnn_msg_kernel<<<(NE + 255) / 256, 256, 0, stream>>>(
        hx, edge_attr, ei, gnn_Wi, gnn_bi, gnn_Wj, gnn_bj, gnn_We, gnn_be,
        gnn_m1W, gnn_m1b, gnn_m2W, gnn_m2b, gnn_lng, gnn_lnb, l, mbuf);
    EZGNN_gnn_aggr_kernel<<<(NN * 16 + 255) / 256, 256, 0, stream>>>(
        mbuf, ei, hx);
  }

  EZGNN_head_kernel<<<1, 64, 0, stream>>>(
      hx, head_lng, head_lnb, head_W1, head_b1, head_W2, head_b2,
      (float*)d_out);
}
